// KGLinkPredictor_85521388798577
// MI455X (gfx1250) — matrix operand order repaired and checked by reading, not compiled, not run on hardware
//
#include <hip/hip_runtime.h>

#define NN     100000
#define EE     1600000
#define QQ     100000
#define FDIM   128      // HEADS*HID
#define NHEADS 4
#define HC     32

typedef __attribute__((ext_vector_type(16))) __bf16 v16bf;
typedef __attribute__((ext_vector_type(8)))  float  v8f;

// ---- bf16 helpers (truncating conversion: cheap, maps to v_perm / d16_hi) ---
__device__ __forceinline__ unsigned int f2bf2(float a, float b) {
    // lo16 = hi16(a), hi16 = hi16(b)  -> single byte-permute on CDNA5
    return (__float_as_uint(a) >> 16) | (__float_as_uint(b) & 0xFFFF0000u);
}
__device__ __forceinline__ unsigned short f2bf1(float a) {
    return (unsigned short)(__float_as_uint(a) >> 16);   // store as d16_hi
}
__device__ __forceinline__ float bf2f(unsigned short u) {
    return __uint_as_float((unsigned int)u << 16);
}

__device__ __forceinline__ void atomicMaxF(float* addr, float val) {
    if (val >= 0.f) atomicMax((int*)addr, __float_as_int(val));
    else            atomicMin((unsigned int*)addr, __float_as_uint(val));
}

// ---------------------------------------------------------------------------
// Hbf[n,0:128] = X[n,0:128] @ W[128,128]  (bf16 WMMA, f32 accum, bf16 out)
// Block = 256 threads = 8 waves; each wave computes a 16-row x 128-col strip.
// W is staged into LDS pre-swizzled into WMMA-B fragment order.
// ---------------------------------------------------------------------------
__global__ __launch_bounds__(256)
void gemm_wmma_kernel(const float* __restrict__ X, const float* __restrict__ W,
                      unsigned short* __restrict__ H, int nRows) {
    __shared__ unsigned short sW[FDIM * FDIM];
    const int tid = threadIdx.x;

    // Swizzle W (row-major [K=128][N=128]) into B-fragment layout:
    // frag (kt,ntile): lane = hi*16+n holds K = kt*32 + {8*hi+e, 16+8*hi+(e-8)}, N = ntile*16+n
    for (int i = tid; i < FDIM * FDIM; i += 256) {
        int kk = i >> 7, nn = i & 127;
        int kt = kk >> 5, kw = kk & 31;
        int hi = (kw >> 3) & 1, e  = (kw & 7) | ((kw >> 4) << 3);
        int ntile = nn >> 4, n = nn & 15;
        int lane = hi * 16 + n;
        sW[(((kt * 8 + ntile) * 32 + lane) << 4) + e] = f2bf1(W[i]);
    }
    __syncthreads();

    const int lane = tid & 31;
    const int wave = tid >> 5;
    const int hi   = lane >> 4;
    const int ml   = lane & 15;
    const int rowBase = blockIdx.x * 128 + wave * 16;

    v8f acc[8] = {};   // 8 N-tiles of 16x16 f32

    int nodeA = rowBase + ml;
    if (nodeA >= nRows) nodeA = nRows - 1;          // clamp: keep EXEC all-1s
    const float* xrow = X + (size_t)nodeA * FDIM;

#pragma unroll
    for (int kt = 0; kt < 4; ++kt) {
        // A fragment: lane holds row ml, K = kt*32 + {8*hi+e, 16+8*hi+e}
        union { v16bf v; unsigned int u[8]; } afrag;
        float t0[8], t1[8];
        const float4* p0 = (const float4*)(xrow + kt * 32 + 8 * hi);
        const float4* p1 = (const float4*)(xrow + kt * 32 + 16 + 8 * hi);
        *(float4*)(&t0[0]) = p0[0]; *(float4*)(&t0[4]) = p0[1];
        *(float4*)(&t1[0]) = p1[0]; *(float4*)(&t1[4]) = p1[1];
#pragma unroll
        for (int e = 0; e < 4; ++e) {
            afrag.u[e]     = f2bf2(t0[2 * e], t0[2 * e + 1]);
            afrag.u[e + 4] = f2bf2(t1[2 * e], t1[2 * e + 1]);
        }
#pragma unroll
        for (int nt = 0; nt < 8; ++nt) {
            v16bf b = *(const v16bf*)(sW + (((kt * 8 + nt) * 32 + lane) << 4));
            acc[nt] = __builtin_amdgcn_wmma_f32_16x16x32_bf16(
                false, afrag.v, false, b, (short)0, acc[nt], false, false);
        }
    }

    // D layout: VGPR r, lanes 0-15 -> M=r, N=lane; lanes 16-31 -> M=r+8
#pragma unroll
    for (int r = 0; r < 8; ++r) {
        int nodeO = rowBase + r + 8 * hi;
        if (nodeO < nRows) {
            unsigned short* hrow = H + (size_t)nodeO * FDIM + ml;
#pragma unroll
            for (int nt = 0; nt < 8; ++nt) hrow[nt * 16] = f2bf1(acc[nt][r]);
        }
    }
}

// a_s[n,h] = sum_c h[n,h*32+c]*att_s[h,c] ; same for a_d   (h is bf16)
__global__ __launch_bounds__(256)
void att_kernel(const unsigned short* __restrict__ H, const float* __restrict__ att_s,
                const float* __restrict__ att_d, float* __restrict__ as,
                float* __restrict__ ad) {
    int t = blockIdx.x * 256 + threadIdx.x;
    if (t >= NN * NHEADS) return;
    int node = t >> 2, hd = t & 3;
    const unsigned short* hp = H + (size_t)node * FDIM + hd * HC;
    const float* ps = att_s + hd * HC;
    const float* pd = att_d + hd * HC;
    float ss = 0.f, dd = 0.f;
#pragma unroll
    for (int c = 0; c < HC; ++c) { float v = bf2f(hp[c]); ss += v * ps[c]; dd += v * pd[c]; }
    as[t] = ss; ad[t] = dd;
}

__global__ __launch_bounds__(256)
void init_kernel(float* __restrict__ m, float* __restrict__ s, float* __restrict__ u) {
    int t = blockIdx.x * 256 + threadIdx.x;
    if (t < NN * FDIM) u[t] = 0.f;
    if (t < NN * NHEADS) { m[t] = -1e30f; s[t] = 0.f; }
}

// pass 1: e = leaky_relu(a_s[src]+a_d[dst]); segment-max into m[dst]
__global__ __launch_bounds__(256)
void edge_max_kernel(const int* __restrict__ src, const int* __restrict__ dst,
                     const float* __restrict__ as, const float* __restrict__ ad,
                     float* __restrict__ ebuf, float* __restrict__ m) {
    int t = blockIdx.x * 256 + threadIdx.x;
    if (t >= EE * NHEADS) return;
    int e = t >> 2, hd = t & 3;
    int sN = src[e], dN = dst[e];
    float v = as[sN * 4 + hd] + ad[dN * 4 + hd];
    v = v > 0.f ? v : 0.2f * v;
    ebuf[t] = v;
    atomicMaxF(&m[dN * 4 + hd], v);
}

// pass 2 (fused): ex = exp(e-m[dst]); s[dst]+=ex; u[dst] += ex*h[src]
// one wave32 per edge; lane l owns channel l of every head
__global__ __launch_bounds__(256)
void edge_accum_kernel(const int* __restrict__ src, const int* __restrict__ dst,
                       const float* __restrict__ ebuf, const float* __restrict__ m,
                       const unsigned short* __restrict__ H, float* __restrict__ s,
                       float* __restrict__ u) {
    int wave = (blockIdx.x * 256 + threadIdx.x) >> 5;
    int lane = threadIdx.x & 31;
    if (wave >= EE) return;
    int sN = src[wave], dN = dst[wave];
    float ex0 = __expf(ebuf[wave * 4 + 0] - m[dN * 4 + 0]);
    float ex1 = __expf(ebuf[wave * 4 + 1] - m[dN * 4 + 1]);
    float ex2 = __expf(ebuf[wave * 4 + 2] - m[dN * 4 + 2]);
    float ex3 = __expf(ebuf[wave * 4 + 3] - m[dN * 4 + 3]);
    if (lane < 4) {
        float exl = (lane == 0) ? ex0 : (lane == 1) ? ex1 : (lane == 2) ? ex2 : ex3;
        atomicAdd(&s[dN * 4 + lane], exl);
    }
    const unsigned short* hs = H + (size_t)sN * FDIM + lane;
    float*                ud = u + (size_t)dN * FDIM + lane;
    atomicAdd(ud + 0 * HC, ex0 * bf2f(hs[0 * HC]));
    atomicAdd(ud + 1 * HC, ex1 * bf2f(hs[1 * HC]));
    atomicAdd(ud + 2 * HC, ex2 * bf2f(hs[2 * HC]));
    atomicAdd(ud + 3 * HC, ex3 * bf2f(hs[3 * HC]));
}

// out = u/(s+eps) + b  [; SiLU]   (in-place over u is fine)
__global__ __launch_bounds__(256)
void finalize_kernel(const float* __restrict__ u, const float* __restrict__ s,
                     const float* __restrict__ b, float* __restrict__ out, int doSilu) {
    int t = blockIdx.x * 256 + threadIdx.x;
    if (t >= NN * FDIM) return;
    int node = t >> 7, c = t & 127, hd = c >> 5;
    float denom = s[node * 4 + hd] + 1e-16f;
    float v = u[t] / denom + b[c];
    if (doSilu) v = v * (1.f / (1.f + __expf(-v)));
    out[t] = v;
}

// DistMult: sigmoid(sum(emb[h]*rel[r]*emb[t])) — one wave32 per query
__global__ __launch_bounds__(256)
void decoder_kernel(const float* __restrict__ emb, const float* __restrict__ rel,
                    const int* __restrict__ hq, const int* __restrict__ rq,
                    const int* __restrict__ tq, float* __restrict__ out) {
    int q = (blockIdx.x * 256 + threadIdx.x) >> 5;
    int lane = threadIdx.x & 31;
    if (q >= QQ) return;
    size_t hI = (size_t)hq[q] * FDIM, tI = (size_t)tq[q] * FDIM, rI = (size_t)rq[q] * FDIM;
    float p = 0.f;
#pragma unroll
    for (int k = 0; k < 4; ++k) {
        int c = k * HC + lane;
        p += emb[hI + c] * rel[rI + c] * emb[tI + c];
    }
#pragma unroll
    for (int off = 16; off; off >>= 1) p += __shfl_xor(p, off, 32);
    if (lane == 0) out[q] = 1.f / (1.f + __expf(-p));
}

extern "C" void kernel_launch(void* const* d_in, const int* in_sizes, int n_in,
                              void* d_out, int out_size, void* d_ws, size_t ws_size,
                              hipStream_t stream) {
    const float* x    = (const float*)d_in[0];
    const int*   esrc = (const int*)d_in[1];
    const int*   edst = (const int*)d_in[2];
    const int*   hq   = (const int*)d_in[3];
    const int*   rq   = (const int*)d_in[4];
    const int*   tq   = (const int*)d_in[5];
    const float* W1   = (const float*)d_in[6];
    const float* b1   = (const float*)d_in[7];
    const float* as1  = (const float*)d_in[8];
    const float* ad1  = (const float*)d_in[9];
    const float* W2   = (const float*)d_in[10];
    const float* b2   = (const float*)d_in[11];
    const float* as2  = (const float*)d_in[12];
    const float* ad2  = (const float*)d_in[13];
    const float* rel  = (const float*)d_in[14];
    float* out = (float*)d_out;

    float* fws  = (float*)d_ws;
    float* bufU = fws;                                 // N*128 f32 (accum / layer out)
    float* aS   = bufU + (size_t)NN * FDIM;            // N*4
    float* aD   = aS + (size_t)NN * NHEADS;            // N*4
    float* mM   = aD + (size_t)NN * NHEADS;            // N*4
    float* sS   = mM + (size_t)NN * NHEADS;            // N*4
    float* eB   = sS + (size_t)NN * NHEADS;            // E*4
    unsigned short* bufH = (unsigned short*)(eB + (size_t)EE * NHEADS);  // N*128 bf16

    const int gGemm  = (NN + 127) / 128;
    const int gAtt   = (NN * NHEADS + 255) / 256;
    const int gInit  = (NN * FDIM + 255) / 256;
    const int gEdge4 = (EE * NHEADS + 255) / 256;
    const int gEdgeW = (EE + 7) / 8;
    const int gFin   = (NN * FDIM + 255) / 256;
    const int gDec   = (QQ + 7) / 8;

    // ---- layer 1 ----
    gemm_wmma_kernel<<<gGemm, 256, 0, stream>>>(x, W1, bufH, NN);
    att_kernel<<<gAtt, 256, 0, stream>>>(bufH, as1, ad1, aS, aD);
    init_kernel<<<gInit, 256, 0, stream>>>(mM, sS, bufU);
    edge_max_kernel<<<gEdge4, 256, 0, stream>>>(esrc, edst, aS, aD, eB, mM);
    edge_accum_kernel<<<gEdgeW, 256, 0, stream>>>(esrc, edst, eB, mM, bufH, sS, bufU);
    finalize_kernel<<<gFin, 256, 0, stream>>>(bufU, sS, b1, bufU, 1);   // + SiLU

    // ---- layer 2 ----
    gemm_wmma_kernel<<<gGemm, 256, 0, stream>>>(bufU, W2, bufH, NN);
    att_kernel<<<gAtt, 256, 0, stream>>>(bufH, as2, ad2, aS, aD);
    init_kernel<<<gInit, 256, 0, stream>>>(mM, sS, bufU);
    edge_max_kernel<<<gEdge4, 256, 0, stream>>>(esrc, edst, aS, aD, eB, mM);
    edge_accum_kernel<<<gEdgeW, 256, 0, stream>>>(esrc, edst, eB, mM, bufH, sS, bufU);
    finalize_kernel<<<gFin, 256, 0, stream>>>(bufU, sS, b2, bufU, 0);   // node_emb

    // ---- decoder ----
    decoder_kernel<<<gDec, 256, 0, stream>>>(bufU, rel, hq, rq, tq, out);
}
